// RegionPooler_33079838113841
// MI455X (gfx1250) — matrix operand order repaired and checked by reading, not compiled, not run on hardware
//
#include <hip/hip_runtime.h>
#include <hip/hip_bf16.h>

#define BB 16
#define TT 512
#define PP 4096
#define DD 768

typedef __attribute__((ext_vector_type(16))) __bf16 v16bf;
typedef __attribute__((ext_vector_type(8)))  __bf16 v8bf;
typedef __attribute__((ext_vector_type(4)))  __bf16 v4bf;
typedef __attribute__((ext_vector_type(8)))  float  v8f;

#if __has_builtin(__builtin_amdgcn_global_load_async_to_lds_b128)
#define ASYNC_G2L 1
// exact parameter type from hipcc diagnostic: int __vector(4) *
typedef int v4i_vs __attribute__((vector_size(16)));
#endif

// Branchless select: keep `a` iff bit==1, else +0.0f (no divergent branch).
__device__ __forceinline__ float selbit(float a, unsigned bit) {
  return __uint_as_float(__float_as_uint(a) & (0u - bit));
}

// 16-byte global -> LDS copy; async-DMA (ASYNCcnt) when the builtin exists.
__device__ __forceinline__ void copy16B_g2l(__bf16* ldst,
                                            const unsigned short* gsrc) {
#ifdef ASYNC_G2L
  __builtin_amdgcn_global_load_async_to_lds_b128(
      (v4i_vs*)(void*)gsrc, (v4i_vs*)(void*)ldst, 0, 0);
#else
  *(v8bf*)ldst = *(const v8bf*)gsrc;
#endif
}

__device__ __forceinline__ void wait_async_g2l() {
#ifdef ASYNC_G2L
#if __has_builtin(__builtin_amdgcn_s_wait_asynccnt)
  __builtin_amdgcn_s_wait_asynccnt(0);
#else
  asm volatile("s_wait_asynccnt 0" ::: "memory");
#endif
#endif
}

// ---------------------------------------------------------------------------
// Kernel 1: E[b,p] = exp(patch_feats[b,p,:] . w + b0); one wave per row.
// Fast path also emits a bf16 copy of patch_feats (coalesced float4 -> 4xbf16).
// ---------------------------------------------------------------------------
__global__ void score_exp_cvt_kernel(const float* __restrict__ pf,
                                     const float* __restrict__ w,
                                     const float* __restrict__ b0,
                                     float* __restrict__ E,
                                     unsigned short* __restrict__ pfb) {
  __shared__ float sw[DD];
  const int tid = threadIdx.x;
  sw[tid]       = w[tid];
  sw[tid + 256] = w[tid + 256];
  sw[tid + 512] = w[tid + 512];
  __syncthreads();

  const int row  = blockIdx.x * 8 + (tid >> 5);
  const int lane = tid & 31;
  const float* r = pf + (size_t)row * DD;
  float s = 0.f;
#pragma unroll
  for (int d = 0; d < DD; d += 32) s = fmaf(r[d + lane], sw[d + lane], s);
#pragma unroll
  for (int off = 16; off > 0; off >>= 1) s += __shfl_xor(s, off, 32);
  if (lane == 0) E[row] = __expf(s + b0[0]);

  if (pfb) {  // bf16 shadow of this block's 8x768 region
    const float4* src4 = (const float4*)(pf + (size_t)blockIdx.x * 8 * DD);
    unsigned short* dst = pfb + (size_t)blockIdx.x * 8 * DD;
#pragma unroll
    for (int i = 0; i < 6; ++i) {
      const int s4 = tid + i * 256;
      const float4 v = src4[s4];
      const v4bf hv = {(__bf16)v.x, (__bf16)v.y, (__bf16)v.z, (__bf16)v.w};
      *(v4bf*)(dst + s4 * 4) = hv;
    }
  }
}

// ---------------------------------------------------------------------------
// Kernel 2: per (b,t) row: in-box bitmask (ballot, 1 DWORD / 32 patches),
// S = sum_{in-box} E, rowScale = any ? 1/S : 0, region_mask output.
// ---------------------------------------------------------------------------
__global__ void rowstat_kernel(const float* __restrict__ tboxg,
                               const float* __restrict__ pboxg,
                               const int*   __restrict__ tmask,
                               const float* __restrict__ E,
                               float* __restrict__ rowScale,
                               float* __restrict__ maskOut,
                               unsigned* __restrict__ maskbits) {
  const int idx  = blockIdx.x * 8 + (threadIdx.x >> 5);
  const int lane = threadIdx.x & 31;
  const int b    = idx >> 9;

  const float4 tb = ((const float4*)tboxg)[idx];
  const int    mk = tmask[idx];
  const float4* pb = ((const float4*)pboxg) + (size_t)b * PP;
  const float*  Eb = E + (size_t)b * PP;
  unsigned* mrow = maskbits + (size_t)idx * (PP / 32);

  float S = 0.f;
  unsigned any = 0;
  for (int c = 0; c < PP / 32; ++c) {
    const int p = c * 32 + lane;
    const float4 bx = pb[p];
    bool in = (bx.x >= tb.x) & (bx.y >= tb.y) & (bx.z <= tb.z) & (bx.w <= tb.w);
    in = in && (mk != 0);
    const unsigned mbits = (unsigned)__ballot(in);  // wave-uniform
    S += in ? Eb[p] : 0.f;
    any |= mbits;
    if (lane == 0) mrow[c] = mbits;
  }
#pragma unroll
  for (int off = 16; off > 0; off >>= 1) S += __shfl_xor(S, off, 32);
  if (lane == 0) {
    rowScale[idx] = any ? (1.f / S) : 0.f;
    maskOut[idx]  = any ? 1.f : 0.f;
  }
}

// ---------------------------------------------------------------------------
// Kernel 3a (fast path): attn[b,t,p] = bit ? E[b,p]*rowScale[b,t] : 0 in bf16.
// One block per (b,t) row; thread handles 16 consecutive patches (32B store).
// ---------------------------------------------------------------------------
__global__ void attn_kernel(const float* __restrict__ E,
                            const float* __restrict__ rowScale,
                            const unsigned* __restrict__ maskbits,
                            unsigned short* __restrict__ attn) {
  const int idx = blockIdx.x;  // b*TT + t
  const int tid = threadIdx.x;
  const int b   = idx >> 9;
  const float wv = rowScale[idx];
  const int p0 = tid * 16;
  const unsigned m =
      maskbits[(size_t)idx * (PP / 32) + (tid >> 1)] >> ((tid & 1) * 16);
  const float4* e4 = (const float4*)(E + (size_t)b * PP + p0);
  float ev[16];
#pragma unroll
  for (int i = 0; i < 4; ++i) {
    const float4 v = e4[i];
    ev[i * 4] = v.x; ev[i * 4 + 1] = v.y; ev[i * 4 + 2] = v.z; ev[i * 4 + 3] = v.w;
  }
  v8bf lo, hi;
#pragma unroll
  for (int j = 0; j < 8; ++j) lo[j] = (__bf16)selbit(ev[j] * wv, (m >> j) & 1u);
#pragma unroll
  for (int j = 0; j < 8; ++j)
    hi[j] = (__bf16)selbit(ev[8 + j] * wv, (m >> (8 + j)) & 1u);
  unsigned short* dst = attn + (size_t)idx * PP + p0;
  *(v8bf*)dst = lo;
  *(v8bf*)(dst + 8) = hi;
}

// ---------------------------------------------------------------------------
// Kernel 3 (fast): pure bf16 GEMM, software-pipelined.
//  - B tile double-buffered in LDS, filled by async DMA (ASYNCcnt) for chunk
//    c+1 while chunk c's WMMAs run; one s_wait_asynccnt + one barrier / chunk.
//  - B fragments preloaded in groups of 4 (one dscnt wait per 4 WMMAs) to
//    stay within VGPR budget; launch_bounds(256,1) so nothing spills.
//  - A fragments straight from global (row-major already matches lane layout).
// ---------------------------------------------------------------------------
__global__ void __launch_bounds__(256, 1)
gemm_fast_kernel(const unsigned short* __restrict__ attn,
                 const unsigned short* __restrict__ pfb,
                 float* __restrict__ out) {
  __shared__ alignas(16) __bf16 Blds[2][32 * 128];
  const int tid = threadIdx.x;
  const int bx = blockIdx.x;
  const int b = bx / 24, rem = bx % 24, mb = rem / 6, nb = rem % 6;
  const int m0 = mb * 128, d0 = nb * 128;
  const int lane = tid & 31, wv = tid >> 5;

  const unsigned short* Asrc = attn + ((size_t)(b * TT + m0)) * PP;
  const unsigned short* Bsrc = pfb + ((size_t)b * PP) * DD + d0;
  const unsigned short* Arow =
      Asrc + (size_t)(wv * 16 + (lane & 15)) * PP + ((lane >> 4) * 8);

  v8f acc[8];
  const v8f vz = {0.f, 0.f, 0.f, 0.f, 0.f, 0.f, 0.f, 0.f};
#pragma unroll
  for (int t = 0; t < 8; ++t) acc[t] = vz;

  // prologue: stage chunk 0 into buffer 0
#pragma unroll
  for (int i = 0; i < 2; ++i) {
    const int s = tid + i * 256;
    copy16B_g2l(&Blds[0][s * 8], Bsrc + (size_t)(s >> 4) * DD + (s & 15) * 8);
  }
  wait_async_g2l();
  __syncthreads();

  for (int c = 0; c < PP / 32; ++c) {
    const int k0 = c * 32;
    const __bf16* cur = Blds[c & 1];

    // issue async fill of next chunk into the other buffer (overlaps compute)
    if (c + 1 < PP / 32) {
      __bf16* nxt = Blds[(c + 1) & 1];
#pragma unroll
      for (int i = 0; i < 2; ++i) {
        const int s = tid + i * 256;
        copy16B_g2l(&nxt[s * 8],
                    Bsrc + (size_t)(k0 + 32 + (s >> 4)) * DD + (s & 15) * 8);
      }
    }

    // A fragment from global: two b128 of this lane's row
    const v8bf alo = *(const v8bf*)(Arow + k0);
    const v8bf ahi = *(const v8bf*)(Arow + k0 + 16);
    const v16bf afrag = __builtin_shufflevector(
        alo, ahi, 0, 1, 2, 3, 4, 5, 6, 7, 8, 9, 10, 11, 12, 13, 14, 15);

    // B fragments in groups of 4: bounded register pressure, amortized waits
    const __bf16* brow = &cur[(lane & 15) + (lane >> 4) * 8 * 128];
#pragma unroll
    for (int g = 0; g < 2; ++g) {
      v16bf bfr[4];
#pragma unroll
      for (int t = 0; t < 4; ++t) {
        const __bf16* bp = brow + (g * 4 + t) * 16;
#pragma unroll
        for (int j = 0; j < 8; ++j) {
          bfr[t][j] = bp[j * 128];
          bfr[t][8 + j] = bp[(16 + j) * 128];
        }
      }
#pragma unroll
      for (int t = 0; t < 4; ++t) {
        acc[g * 4 + t] = __builtin_amdgcn_wmma_f32_16x16x32_bf16(
            false, afrag, false, bfr[t], (short)0, acc[g * 4 + t], false,
            false);
      }
    }

    wait_async_g2l();  // next buffer's DMA done (this wave)
    __syncthreads();   // all waves: reads of cur done, writes of nxt visible
  }

  const int nbase = d0 + (lane & 15);
  const int mbase = m0 + wv * 16 + ((lane >> 4) * 8);
#pragma unroll
  for (int t = 0; t < 8; ++t) {
#pragma unroll
    for (int r = 0; r < 8; ++r) {
      __builtin_nontemporal_store(
          acc[t][r], &out[((size_t)b * TT + mbase + r) * DD + nbase + t * 16]);
    }
  }
}

// ---------------------------------------------------------------------------
// Kernel 3 (fallback, small workspace): fused GEMM. A fragment built per lane
// in registers, branchless, from bitmask + E + rowScale. B tile f32->bf16 LDS.
// ---------------------------------------------------------------------------
__global__ void __launch_bounds__(256, 1)
gemm_fused_kernel(const float* __restrict__ pf,
                  const float* __restrict__ E,
                  const float* __restrict__ rowScale,
                  const unsigned* __restrict__ maskbits,
                  float* __restrict__ out) {
  __shared__ alignas(16) __bf16 Blds[32 * 128];
  const int tid = threadIdx.x;
  const int bx = blockIdx.x;
  const int b = bx / 24, rem = bx % 24, mb = rem / 6, nb = rem % 6;
  const int m0 = mb * 128, d0 = nb * 128;
  const int lane = tid & 31, wv = tid >> 5;

  // loop-invariant per-lane row state (no LDS, no divergence)
  const int rowM = m0 + wv * 16 + (lane & 15);
  const int kb   = (lane >> 4) * 8;
  const float wr = rowScale[(size_t)b * TT + rowM];
  const unsigned* mrow = maskbits + ((size_t)(b * TT + rowM)) * (PP / 32);
  const float* Eb  = E + (size_t)b * PP;
  const float* pfB = pf + (size_t)b * PP * DD + d0;

  v8f acc[8];
  const v8f vz = {0.f, 0.f, 0.f, 0.f, 0.f, 0.f, 0.f, 0.f};
#pragma unroll
  for (int t = 0; t < 8; ++t) acc[t] = vz;

  for (int k0 = 0; k0 < PP; k0 += 32) {
    __syncthreads();  // previous iteration's Blds reads done
    // stage B: f32 -> bf16, float4 per slot, 4 per thread
    {
      const float* src = pfB + (size_t)k0 * DD;
#pragma unroll
      for (int i = 0; i < 4; ++i) {
        const int e4s = tid + i * 256;
        const int rr = e4s >> 5;
        const int c4 = (e4s & 31) * 4;
        const float4 v = *(const float4*)(src + (size_t)rr * DD + c4);
        const v4bf hv = {(__bf16)v.x, (__bf16)v.y, (__bf16)v.z, (__bf16)v.w};
        *(v4bf*)&Blds[rr * 128 + c4] = hv;
      }
    }
    // build A fragment in registers (branchless)
    v16bf afrag;
    {
      const unsigned m = mrow[k0 >> 5];
      const float4* e4a = (const float4*)(Eb + k0 + kb);
      const float4* e4b = (const float4*)(Eb + k0 + kb + 16);
      float ev[16];
      const float4 v0 = e4a[0], v1 = e4a[1], v2 = e4b[0], v3 = e4b[1];
      ev[0]=v0.x; ev[1]=v0.y; ev[2]=v0.z; ev[3]=v0.w;
      ev[4]=v1.x; ev[5]=v1.y; ev[6]=v1.z; ev[7]=v1.w;
      ev[8]=v2.x; ev[9]=v2.y; ev[10]=v2.z; ev[11]=v2.w;
      ev[12]=v3.x; ev[13]=v3.y; ev[14]=v3.z; ev[15]=v3.w;
#pragma unroll
      for (int j = 0; j < 8; ++j)
        afrag[j] = (__bf16)selbit(ev[j] * wr, (m >> (kb + j)) & 1u);
#pragma unroll
      for (int j = 0; j < 8; ++j)
        afrag[8 + j] = (__bf16)selbit(ev[8 + j] * wr, (m >> (kb + 16 + j)) & 1u);
    }
    __syncthreads();

    const __bf16* brow = &Blds[(lane & 15) + (lane >> 4) * 8 * 128];
#pragma unroll
    for (int g = 0; g < 2; ++g) {
      v16bf bfr[4];
#pragma unroll
      for (int t = 0; t < 4; ++t) {
        const __bf16* bp = brow + (g * 4 + t) * 16;
#pragma unroll
        for (int j = 0; j < 8; ++j) {
          bfr[t][j] = bp[j * 128];
          bfr[t][8 + j] = bp[(16 + j) * 128];
        }
      }
#pragma unroll
      for (int t = 0; t < 4; ++t) {
        acc[g * 4 + t] = __builtin_amdgcn_wmma_f32_16x16x32_bf16(
            false, afrag, false, bfr[t], (short)0, acc[g * 4 + t], false,
            false);
      }
    }
  }

  const int nbase = d0 + (lane & 15);
  const int mbase = m0 + wv * 16 + ((lane >> 4) * 8);
#pragma unroll
  for (int t = 0; t < 8; ++t) {
#pragma unroll
    for (int r = 0; r < 8; ++r) {
      __builtin_nontemporal_store(
          acc[t][r], &out[((size_t)b * TT + mbase + r) * DD + nbase + t * 16]);
    }
  }
}

// ---------------------------------------------------------------------------
// Workspace layout
//   [0)        E         BB*PP f32      = 256 KB
//   [256K)     rowScale  BB*TT f32      =  32 KB
//   [288K)     maskbits  BB*TT*128 u32  =   4 MB
//   [8M)       pfb       BB*PP*DD bf16  =  96 MB   (fast path only)
//   [8M+96M)   attn      BB*TT*PP bf16  =  64 MB   (fast path only)
// ---------------------------------------------------------------------------
#define OFF_RS   ((size_t)BB * PP * 4)
#define OFF_MB   (OFF_RS + (size_t)BB * TT * 4)
#define OFF_PFB  ((size_t)8 << 20)
#define OFF_ATTN (OFF_PFB + (size_t)BB * PP * DD * 2)
#define NEED_FAST (OFF_ATTN + (size_t)BB * TT * PP * 2)

extern "C" void kernel_launch(void* const* d_in, const int* in_sizes, int n_in,
                              void* d_out, int out_size, void* d_ws, size_t ws_size,
                              hipStream_t stream) {
  const float* pf    = (const float*)d_in[0];
  const float* tbox  = (const float*)d_in[1];
  const float* pbox  = (const float*)d_in[2];
  const int*   tmask = (const int*)d_in[3];
  const float* w     = (const float*)d_in[4];
  const float* b0    = (const float*)d_in[5];

  float* out     = (float*)d_out;
  float* maskOut = out + (size_t)BB * TT * DD;

  char* ws = (char*)d_ws;
  float*          E        = (float*)ws;
  float*          rowScale = (float*)(ws + OFF_RS);
  unsigned*       maskbits = (unsigned*)(ws + OFF_MB);
  unsigned short* pfb      = (unsigned short*)(ws + OFF_PFB);
  unsigned short* attnW    = (unsigned short*)(ws + OFF_ATTN);

  const bool fast = (ws_size >= NEED_FAST);

  score_exp_cvt_kernel<<<BB * PP / 8, 256, 0, stream>>>(pf, w, b0, E,
                                                        fast ? pfb : nullptr);
  rowstat_kernel<<<BB * TT / 8, 256, 0, stream>>>(tbox, pbox, tmask, E,
                                                  rowScale, maskOut, maskbits);
  if (fast) {
    attn_kernel<<<BB * TT, 256, 0, stream>>>(E, rowScale, maskbits, attnW);
    gemm_fast_kernel<<<BB * 24, 256, 0, stream>>>(attnW, pfb, out);
  } else {
    gemm_fused_kernel<<<BB * 24, 256, 0, stream>>>(pf, E, rowScale, maskbits,
                                                   out);
  }
}
